// PointerGenDecoder_61125974556737
// MI455X (gfx1250) — matrix operand order repaired and checked by reading, not compiled, not run on hardware
//
#include <hip/hip_runtime.h>
#include <hip/hip_bf16.h>

// ---------------------------------------------------------------------------
// Problem constants (from reference): B=16, T=128, S=256, D=512, VOCAB=32000
// ---------------------------------------------------------------------------
#define CB 16
#define CT 128
#define CS 256
#define CD 512
#define CV 32000

typedef __attribute__((ext_vector_type(16))) __bf16 v16bf;
typedef __attribute__((ext_vector_type(8)))  float  v8f;

__device__ __forceinline__ float sigmoidf_(float x) { return 1.0f / (1.0f + __expf(-x)); }

// ---------------------------------------------------------------------------
// Wave-tile bf16 WMMA GEMM:  C[m,n] = sum_k A[m,k]*B[n,k] (+bias0+bias1)(+addsrc)
// A: M x K bf16 row-major;  B: N x K bf16 row-major (computes A @ B^T).
// Each wave computes TWO adjacent 16x16 output tiles (one 16x32 strip):
//  - one shared A fragment per k-step, two independent accumulator chains
//    -> 2x matrix-pipe ILP, half the A-operand loads.
// K stepped by 64 (all K used are multiples of 64) -> 4 WMMAs per iteration,
// no remainder path.  M mult of 16, N mult of 32.
// ---------------------------------------------------------------------------
__global__ __launch_bounds__(256) void gemm_bf16_wmma(
    const __bf16* __restrict__ A, const __bf16* __restrict__ B,
    const float* __restrict__ bias0, const float* __restrict__ bias1,
    const float* __restrict__ addsrc,
    float* __restrict__ Cf, __bf16* __restrict__ Cb,
    int M, int N, int K)
{
  const int wave = (blockIdx.x * blockDim.x + threadIdx.x) >> 5;
  const int lane = threadIdx.x & 31;
  const int ntp  = N >> 5;                   // N-tile PAIRS
  const int total = (M >> 4) * ntp;
  if (wave >= total) return;                 // wave-uniform: EXEC stays all-1s
  const int mt   = wave / ntp;
  const int nt0  = (wave % ntp) * 2;         // first of the two N-tiles
  const int half = lane >> 4;                // 0: lanes 0-15, 1: lanes 16-31
  const int l16  = lane & 15;

  const __bf16* Ar  = A + (size_t)(mt * 16 + l16) * K;   // A row M = mt*16 + l16
  const __bf16* Br0 = B + (size_t)(nt0 * 16 + l16) * K;  // B col N = nt0*16 + l16
  const __bf16* Br1 = Br0 + (size_t)16 * K;              // next N-tile

  union U32B { uint4 u[2]; v16bf v; };
  v8f acc0 = {}, acc1 = {};

  for (int k0 = 0; k0 < K; k0 += 64) {
    U32B a0, a1, b00, b01, b10, b11;
    // A 16x32 operand: elems 0..15 -> k+half*16 .. +15
    a0.u[0] = *(const uint4*)(Ar + k0 + half * 16);
    a0.u[1] = *(const uint4*)(Ar + k0 + half * 16 + 8);
    a1.u[0] = *(const uint4*)(Ar + k0 + 32 + half * 16);
    a1.u[1] = *(const uint4*)(Ar + k0 + 32 + half * 16 + 8);
    // B 32x16 operand: elems 0..15 -> k+half*16 .. +15
    b00.u[0] = *(const uint4*)(Br0 + k0 + half * 16);
    b00.u[1] = *(const uint4*)(Br0 + k0 + half * 16 + 8);
    b10.u[0] = *(const uint4*)(Br1 + k0 + half * 16);
    b10.u[1] = *(const uint4*)(Br1 + k0 + half * 16 + 8);
    b01.u[0] = *(const uint4*)(Br0 + k0 + 32 + half * 16);
    b01.u[1] = *(const uint4*)(Br0 + k0 + 32 + half * 16 + 8);
    b11.u[0] = *(const uint4*)(Br1 + k0 + 32 + half * 16);
    b11.u[1] = *(const uint4*)(Br1 + k0 + 32 + half * 16 + 8);
    acc0 = __builtin_amdgcn_wmma_f32_16x16x32_bf16(false, a0.v, false, b00.v, (short)0, acc0, false, false);
    acc1 = __builtin_amdgcn_wmma_f32_16x16x32_bf16(false, a0.v, false, b10.v, (short)0, acc1, false, false);
    acc0 = __builtin_amdgcn_wmma_f32_16x16x32_bf16(false, a1.v, false, b01.v, (short)0, acc0, false, false);
    acc1 = __builtin_amdgcn_wmma_f32_16x16x32_bf16(false, a1.v, false, b11.v, (short)0, acc1, false, false);
  }

  // epilogue for both tiles: C/D layout: elem r -> M = r + half*8, N = l16
  const int n0 = nt0 * 16 + l16;
  const int n1 = n0 + 16;
  float badd0 = 0.0f, badd1 = 0.0f;
  if (bias0) { badd0 += bias0[n0]; badd1 += bias0[n1]; }
  if (bias1) { badd0 += bias1[n0]; badd1 += bias1[n1]; }
  #pragma unroll
  for (int r = 0; r < 8; ++r) {
    const int m = mt * 16 + r + half * 8;
    float v0 = acc0[r] + badd0;
    float v1 = acc1[r] + badd1;
    if (addsrc) {
      v0 += addsrc[(size_t)m * N + n0];
      v1 += addsrc[(size_t)m * N + n1];
    }
    if (Cf) { Cf[(size_t)m * N + n0] = v0; Cf[(size_t)m * N + n1] = v1; }
    if (Cb) { Cb[(size_t)m * N + n0] = (__bf16)v0; Cb[(size_t)m * N + n1] = (__bf16)v1; }
  }
}

// ---------------------------------------------------------------------------
// fp32 -> bf16 conversion (grid-stride)
// ---------------------------------------------------------------------------
__global__ void f32_to_bf16_k(const float* __restrict__ src, __bf16* __restrict__ dst, int n)
{
  int i = blockIdx.x * blockDim.x + threadIdx.x;
  int stride = gridDim.x * blockDim.x;
  for (; i < n; i += stride) dst[i] = (__bf16)src[i];
}

// ---------------------------------------------------------------------------
// Embedding gather (to bf16, layout row = t*B + b) + xdot[t*B+b] = emb[b,t]·W_xt
// grid = B*T blocks (block id = b*T + t), 256 threads
// ---------------------------------------------------------------------------
__global__ __launch_bounds__(256) void emb_gather_k(
    const int* __restrict__ words, const float* __restrict__ embedding,
    const float* __restrict__ W_xt, __bf16* __restrict__ Xemb, float* __restrict__ xdot)
{
  const int bt = blockIdx.x;
  const int b = bt / CT, t = bt % CT;
  const int w = words[bt];
  const float* e = embedding + (size_t)w * CD;
  __bf16* xe = Xemb + (size_t)(t * CB + b) * CD;
  float partial = 0.0f;
  for (int k = threadIdx.x; k < CD; k += 256) {
    float v = e[k];
    xe[k] = (__bf16)v;
    partial += v * W_xt[k];
  }
  __shared__ float red[256];
  red[threadIdx.x] = partial; __syncthreads();
  for (int s = 128; s > 0; s >>= 1) {
    if (threadIdx.x < s) red[threadIdx.x] += red[threadIdx.x + s];
    __syncthreads();
  }
  if (threadIdx.x == 0) xdot[t * CB + b] = red[0];
}

// ---------------------------------------------------------------------------
// pre_h (S,B,D) fp32 -> preh_bf (b*S+s, D) bf16    grid = B*S blocks
// ---------------------------------------------------------------------------
__global__ __launch_bounds__(256) void preh_gather_k(
    const float* __restrict__ pre_h, __bf16* __restrict__ preh_bf)
{
  const int bs = blockIdx.x;
  const int b = bs / CS, s = bs % CS;
  const float* srcp = pre_h + ((size_t)s * CB + b) * CD;
  __bf16* dst = preh_bf + (size_t)bs * CD;
  for (int k = threadIdx.x; k < CD; k += 256) dst[k] = (__bf16)srcp[k];
}

// ---------------------------------------------------------------------------
// h0 = pre_h[input_len-1, b, :] (bf16), c0 = cell_init    grid = B blocks
// ---------------------------------------------------------------------------
__global__ __launch_bounds__(256) void init_state_k(
    const float* __restrict__ pre_h, const float* __restrict__ cell_init,
    const int* __restrict__ input_len, __bf16* __restrict__ h_bf, float* __restrict__ c)
{
  const int b = blockIdx.x;
  const int len = input_len[b];
  const float* hsrc = pre_h + ((size_t)(len - 1) * CB + b) * CD;
  for (int k = threadIdx.x; k < CD; k += 256) {
    h_bf[b * CD + k] = (__bf16)hsrc[k];
    c[b * CD + k] = cell_init[b * CD + k];
  }
}

// ---------------------------------------------------------------------------
// Fused LSTM cell + attention for one step.  grid = B blocks, 256 threads.
// ---------------------------------------------------------------------------
__global__ __launch_bounds__(256) void cell_attn_k(
    const float* __restrict__ gates,      // [B][4D]
    float* __restrict__ c,                // [B][D] in/out
    __bf16* __restrict__ h_bf,            // [B][D] out
    __bf16* __restrict__ concat_bf,       // [B][2D] out
    const float* __restrict__ pre_h,      // [S][B][D]
    const float* __restrict__ Wh_pre,     // [B*S][D]
    const float* __restrict__ W_s,        // [D][D]
    const float* __restrict__ b_s,        // [D]
    const float* __restrict__ v_t,        // [D]
    const float* __restrict__ W_hp,       // [D]
    const float* __restrict__ W_st,       // [D]
    const float* __restrict__ b_st,       // [1]
    const float* __restrict__ xdot,       // [T][B]
    const int* __restrict__ src,          // [B][S]
    const int* __restrict__ input_len,    // [B]
    const int* __restrict__ words,        // [B][T]
    float* __restrict__ pgen, float* __restrict__ copyv,
    int t)
{
  const int b = blockIdx.x;
  const int tid = threadIdx.x;
  __shared__ float s_c[CD], s_sp[CD], s_ctx[CD], s_a[CS], red[256];

  const float* g = gates + (size_t)b * 4 * CD;
  // 1. LSTM cell
  for (int d = tid; d < CD; d += 256) {
    float ig = g[d], fg = g[CD + d], gg = g[2 * CD + d], og = g[3 * CD + d];
    float cn = sigmoidf_(fg) * c[b * CD + d] + sigmoidf_(ig) * tanhf(gg);
    float hn = sigmoidf_(og) * tanhf(cn);
    s_c[d] = cn;
    c[b * CD + d] = cn;
    h_bf[b * CD + d] = (__bf16)hn;
    concat_bf[b * 2 * CD + CD + d] = (__bf16)cn;
  }
  __syncthreads();

  // 2. s_proj = c_new @ W_s^T + b_s
  for (int d = tid; d < CD; d += 256) {
    const float* wr = W_s + (size_t)d * CD;
    float acc = b_s[d];
    for (int k = 0; k < CD; ++k) acc += s_c[k] * wr[k];
    s_sp[d] = acc;
  }
  __syncthreads();

  // 3. e[s] = tanh(Wh_pre + s_proj)·v_t, masked
  const int len = input_len[b];
  {
    const int s = tid;   // S == 256 == blockDim
    const float* wh = Wh_pre + ((size_t)b * CS + s) * CD;
    float e = 0.0f;
    for (int k = 0; k < CD; ++k) e += tanhf(wh[k] + s_sp[k]) * v_t[k];
    s_a[s] = (s < len) ? e : -1.0e9f;
  }
  __syncthreads();

  // 4. softmax over S
  red[tid] = s_a[tid]; __syncthreads();
  for (int st = 128; st > 0; st >>= 1) {
    if (tid < st) red[tid] = fmaxf(red[tid], red[tid + st]);
    __syncthreads();
  }
  const float mx = red[0]; __syncthreads();
  const float ex = __expf(s_a[tid] - mx);
  red[tid] = ex; __syncthreads();
  for (int st = 128; st > 0; st >>= 1) {
    if (tid < st) red[tid] += red[tid + st];
    __syncthreads();
  }
  const float inv = 1.0f / red[0];
  __syncthreads();
  s_a[tid] = ex * inv;

  // 5. copy prob
  const int wj = words[b * CT + t];
  float cp = ((tid < len) && (src[b * CS + tid] == wj)) ? s_a[tid] : 0.0f;
  __syncthreads();
  red[tid] = cp; __syncthreads();
  for (int st = 128; st > 0; st >>= 1) {
    if (tid < st) red[tid] += red[tid + st];
    __syncthreads();
  }
  const float copy_b = red[0];
  __syncthreads();

  // 6. context = a @ preh   (coalesced over d across threads)
  for (int d = tid; d < CD; d += 256) {
    float acc = 0.0f;
    for (int s = 0; s < CS; ++s)
      acc += s_a[s] * pre_h[((size_t)s * CB + b) * CD + d];
    s_ctx[d] = acc;
    concat_bf[b * 2 * CD + d] = (__bf16)acc;
  }
  __syncthreads();

  // 7. p_gen
  float p = 0.0f;
  for (int d = tid; d < CD; d += 256) p += s_ctx[d] * W_hp[d] + s_c[d] * W_st[d];
  red[tid] = p; __syncthreads();
  for (int st = 128; st > 0; st >>= 1) {
    if (tid < st) red[tid] += red[tid + st];
    __syncthreads();
  }
  if (tid == 0) {
    float z = red[0] + b_st[0] + xdot[(t + 1) * CB + b];
    pgen[b]  = sigmoidf_(z);
    copyv[b] = copy_b;
  }
}

// ---------------------------------------------------------------------------
// Vocab softmax (online max/sum over 32000) + pointer-gen mix -> out[b, t]
// grid = B blocks, 256 threads
// ---------------------------------------------------------------------------
__global__ __launch_bounds__(256) void vocab_out_k(
    const float* __restrict__ logits,     // [B][V]
    const float* __restrict__ pgen, const float* __restrict__ copyv,
    const int* __restrict__ words, const int* __restrict__ lengths,
    float* __restrict__ out, int t)
{
  const int b = blockIdx.x, tid = threadIdx.x;
  const float* row = logits + (size_t)b * CV;
  float mx = -1.0e30f, sm = 0.0f;
  for (int i = tid; i < CV; i += 256) {
    float v = row[i];
    if (v > mx) { sm = sm * __expf(mx - v) + 1.0f; mx = v; }
    else        { sm += __expf(v - mx); }
  }
  __shared__ float rmx[256], rsm[256];
  rmx[tid] = mx; rsm[tid] = sm; __syncthreads();
  for (int st = 128; st > 0; st >>= 1) {
    if (tid < st) {
      float m1 = rmx[tid], s1 = rsm[tid];
      float m2 = rmx[tid + st], s2 = rsm[tid + st];
      float m = fmaxf(m1, m2);
      rmx[tid] = m;
      rsm[tid] = s1 * __expf(m1 - m) + s2 * __expf(m2 - m);
    }
    __syncthreads();
  }
  if (tid == 0) {
    const int wj1 = words[b * CT + t + 1];
    float P = __expf(row[wj1] - rmx[0]) / rsm[0];
    float pg = pgen[b];
    float o = pg * P + (1.0f - pg) * copyv[b];
    const int len = lengths[b];
    out[b * (CT - 1) + t] = (t < len - 1) ? o : 0.0f;
  }
}

// ---------------------------------------------------------------------------
// Host launcher
// ---------------------------------------------------------------------------
extern "C" void kernel_launch(void* const* d_in, const int* in_sizes, int n_in,
                              void* d_out, int out_size, void* d_ws, size_t ws_size,
                              hipStream_t stream)
{
  (void)in_sizes; (void)n_in; (void)out_size; (void)ws_size;
  const int*   words     = (const int*)  d_in[0];
  const int*   lengths   = (const int*)  d_in[1];
  const int*   src       = (const int*)  d_in[2];
  const int*   input_len = (const int*)  d_in[3];
  const float* pre_h     = (const float*)d_in[4];
  const float* cell_init = (const float*)d_in[5];
  const float* embedding = (const float*)d_in[6];
  const float* W_ih      = (const float*)d_in[7];
  const float* W_hh      = (const float*)d_in[8];
  const float* b_ih      = (const float*)d_in[9];
  const float* b_hh      = (const float*)d_in[10];
  const float* W_h       = (const float*)d_in[11];
  const float* W_hp      = (const float*)d_in[12];
  const float* W_s       = (const float*)d_in[13];
  const float* b_s       = (const float*)d_in[14];
  const float* W_st      = (const float*)d_in[15];
  const float* b_st      = (const float*)d_in[16];
  const float* W_xt      = (const float*)d_in[17];
  const float* v_t       = (const float*)d_in[18];
  const float* V_w       = (const float*)d_in[19];
  const float* V_b       = (const float*)d_in[20];
  const float* Vp_w      = (const float*)d_in[21];
  const float* Vp_b      = (const float*)d_in[22];
  float* out = (float*)d_out;

  // workspace carve (256B aligned)
  char* p = (char*)d_ws;
  auto carve = [&](size_t bytes) -> char* {
    char* r = p;
    p += (bytes + 255) & ~(size_t)255;
    return r;
  };
  __bf16* Vp_bf   = (__bf16*)carve((size_t)CV * CD * 2);          // 32.8 MB
  __bf16* Whh_bf  = (__bf16*)carve((size_t)4 * CD * CD * 2);      //  2.1 MB
  __bf16* Wih_bf  = (__bf16*)carve((size_t)4 * CD * CD * 2);      //  2.1 MB
  __bf16* Wh_bf   = (__bf16*)carve((size_t)CD * CD * 2);          //  0.5 MB
  __bf16* Vw_bf   = (__bf16*)carve((size_t)CD * 2 * CD * 2);      //  1.0 MB
  __bf16* Xemb    = (__bf16*)carve((size_t)CT * CB * CD * 2);     //  2.1 MB
  __bf16* preh_bf = (__bf16*)carve((size_t)CB * CS * CD * 2);     //  4.2 MB
  float*  WhPre   = (float*) carve((size_t)CB * CS * CD * 4);     //  8.4 MB
  float*  Gx      = (float*) carve((size_t)(CT - 1) * CB * 4 * CD * 4); // 16.6 MB
  float*  gatesWS = (float*) carve((size_t)CB * 4 * CD * 4);
  float*  cbuf    = (float*) carve((size_t)CB * CD * 4);
  __bf16* h_bf    = (__bf16*)carve((size_t)CB * CD * 2);
  __bf16* cat_bf  = (__bf16*)carve((size_t)CB * 2 * CD * 2);
  __bf16* hid_bf  = (__bf16*)carve((size_t)CB * CD * 2);
  float*  logits  = (float*) carve((size_t)CB * CV * 4);          //  2.0 MB
  float*  xdot    = (float*) carve((size_t)CT * CB * 4);
  float*  pgen    = (float*) carve((size_t)CB * 4);
  float*  copyv   = (float*) carve((size_t)CB * 4);

  // ---- phase 0: precompute (parallel, recurrence-free) ----
  f32_to_bf16_k<<<4096, 256, 0, stream>>>(Vp_w,  Vp_bf,  CV * CD);
  f32_to_bf16_k<<<1024, 256, 0, stream>>>(W_hh,  Whh_bf, 4 * CD * CD);
  f32_to_bf16_k<<<1024, 256, 0, stream>>>(W_ih,  Wih_bf, 4 * CD * CD);
  f32_to_bf16_k<<< 256, 256, 0, stream>>>(W_h,   Wh_bf,  CD * CD);
  f32_to_bf16_k<<< 512, 256, 0, stream>>>(V_w,   Vw_bf,  CD * 2 * CD);
  emb_gather_k <<<CB * CT, 256, 0, stream>>>(words, embedding, W_xt, Xemb, xdot);
  preh_gather_k<<<CB * CS, 256, 0, stream>>>(pre_h, preh_bf);
  init_state_k <<<CB, 256, 0, stream>>>(pre_h, cell_init, input_len, h_bf, cbuf);

  {   // Gx[t*B+b, :] = x_t @ W_ih^T + (b_ih + b_hh)     M=2032, N=2048, K=512
    const int M = (CT - 1) * CB, N = 4 * CD, K = CD;
    const int total = (M / 16) * (N / 32);
    gemm_bf16_wmma<<<(total + 7) / 8, 256, 0, stream>>>(
        Xemb, Wih_bf, b_ih, b_hh, nullptr, Gx, nullptr, M, N, K);
  }
  {   // Wh_pre = preh @ W_h^T                           M=4096, N=512, K=512
    const int M = CB * CS, N = CD, K = CD;
    const int total = (M / 16) * (N / 32);
    gemm_bf16_wmma<<<(total + 7) / 8, 256, 0, stream>>>(
        preh_bf, Wh_bf, nullptr, nullptr, nullptr, WhPre, nullptr, M, N, K);
  }

  // ---- phase 1: sequential scan over T-1 steps ----
  for (int t = 0; t < CT - 1; ++t) {
    {   // gates = h @ W_hh^T + Gx[t]                    M=16, N=2048, K=512
      const int M = CB, N = 4 * CD, K = CD;
      const int total = (M / 16) * (N / 32);
      gemm_bf16_wmma<<<(total + 7) / 8, 256, 0, stream>>>(
          h_bf, Whh_bf, nullptr, nullptr, Gx + (size_t)t * CB * 4 * CD,
          gatesWS, nullptr, M, N, K);
    }
    cell_attn_k<<<CB, 256, 0, stream>>>(
        gatesWS, cbuf, h_bf, cat_bf, pre_h, WhPre, W_s, b_s, v_t,
        W_hp, W_st, b_st, xdot, src, input_len, words, pgen, copyv, t);
    {   // hidden = [context, c] @ V_w^T + V_b (bf16)    M=16, N=512, K=1024
      const int M = CB, N = CD, K = 2 * CD;
      const int total = (M / 16) * (N / 32);
      gemm_bf16_wmma<<<(total + 7) / 8, 256, 0, stream>>>(
          cat_bf, Vw_bf, V_b, nullptr, nullptr, nullptr, hid_bf, M, N, K);
    }
    {   // logits = hidden @ Vp_w^T + Vp_b               M=16, N=32000, K=512
      const int M = CB, N = CV, K = CD;
      const int total = (M / 16) * (N / 32);
      gemm_bf16_wmma<<<(total + 7) / 8, 256, 0, stream>>>(
          hid_bf, Vp_bf, Vp_b, nullptr, nullptr, logits, nullptr, M, N, K);
    }
    vocab_out_k<<<CB, 256, 0, stream>>>(logits, pgen, copyv, words, lengths, out, t);
  }
}
